// Model_33827162423446
// MI455X (gfx1250) — matrix operand order repaired and checked by reading, not compiled, not run on hardware
//
#include <hip/hip_runtime.h>
#include <math.h>

// ---------------- model constants ----------------
static const int B_ = 16, L_ = 512, N_ = 321, MARK_ = 4;
static const int T_ = N_ + MARK_;          // 325 tokens
static const int D_ = 512, S_ = 16, R_ = 32, F_ = 2048, EL_ = 3, P_ = 96;
static const int M_ = B_ * T_;             // 5200 rows for all GEMMs

typedef _Float16 h16 __attribute__((ext_vector_type(16)));
typedef _Float16 h4  __attribute__((ext_vector_type(4)));
typedef float    f8  __attribute__((ext_vector_type(8)));
typedef float    f4  __attribute__((ext_vector_type(4)));

// =====================================================================
// WMMA GEMM:  C[M,Nn] = act( A[M,K](lda) * W[Nn,K]^T + bias )
// block = 256 threads = 8 wave32; block tile 64(M) x 64(N); K step 32.
// Each wave computes 32 rows x 16 cols = 2 WMMAs sharing one B fragment.
// Double-buffered LDS + software pipeline: one barrier per K-step,
// global loads for tile k+32 overlap the WMMAs on tile k.
// LDS rows padded to 40 halves (80B) -> fragment-read rows start at
// banks spaced 4 apart (minimum 2-way conflicts), 16B alignment kept.
// act: 0 = none, 1 = relu, 2 = softplus
// =====================================================================
__global__ __launch_bounds__(256) void wmma_gemm_kernel(
    const float* __restrict__ A, int lda,
    const float* __restrict__ W,            // (Nn, K) row-major
    const float* __restrict__ bias,         // may be null
    float* __restrict__ C, int ldc,
    int M, int Nn, int K, int act)
{
    const int TK = 32;
    const int PAD = 40;                     // 32 + 8 halves
    __shared__ _Float16 As[2][64][PAD];
    __shared__ _Float16 Bs[2][64][PAD];

    const int m0   = blockIdx.y * 64;
    const int n0   = blockIdx.x * 64;
    const int tid  = threadIdx.x;
    const int lane = tid & 31;
    const int wave = tid >> 5;      // 0..7
    const int wr   = wave >> 2;     // 0..1  -> 32-row band
    const int wc   = wave & 3;      // 0..3  -> 16-col sub-tile
    const int lm   = lane & 15;
    const int hi   = lane >> 4;     // 0 or 1

    // staging mapping: 64x32 tile = 2048 floats; 8 floats/thread as 2 float4
    const int sr = tid >> 3;          // 0..31 (rows; +32 for second chunk)
    const int sc = (tid & 7) * 4;     // 0,4,...,28

    // loop-invariant predicates + streaming pointers
    const bool okA0 = (m0 + sr)      < M;
    const bool okA1 = (m0 + sr + 32) < M;
    const bool okB0 = (n0 + sr)      < Nn;
    const bool okB1 = (n0 + sr + 32) < Nn;
    const float* pa0 = A + (size_t)(m0 + sr)      * lda + sc;
    const float* pa1 = A + (size_t)(m0 + sr + 32) * lda + sc;
    const float* pb0 = W + (size_t)(n0 + sr)      * K   + sc;
    const float* pb1 = W + (size_t)(n0 + sr + 32) * K   + sc;

    f8 acc0 = {0.f, 0.f, 0.f, 0.f, 0.f, 0.f, 0.f, 0.f};
    f8 acc1 = {0.f, 0.f, 0.f, 0.f, 0.f, 0.f, 0.f, 0.f};

    const f4 zero4 = {0.f, 0.f, 0.f, 0.f};

    // ---- prologue: stage tile k0=0 into buffer 0 ----
    {
        f4 ra0 = zero4, ra1 = zero4, rb0 = zero4, rb1 = zero4;
        if (okA0) ra0 = *(const f4*)pa0;
        if (okA1) ra1 = *(const f4*)pa1;
        if (okB0) rb0 = *(const f4*)pb0;
        if (okB1) rb1 = *(const f4*)pb1;
        *(h4*)&As[0][sr][sc]      = __builtin_convertvector(ra0, h4);
        *(h4*)&As[0][sr + 32][sc] = __builtin_convertvector(ra1, h4);
        *(h4*)&Bs[0][sr][sc]      = __builtin_convertvector(rb0, h4);
        *(h4*)&Bs[0][sr + 32][sc] = __builtin_convertvector(rb1, h4);
    }
    __syncthreads();

    int cur = 0;
    for (int k0 = 0; k0 < K; k0 += TK) {
        const bool has_next = (k0 + TK < K);

        // issue global loads for the NEXT tile (overlaps with WMMA below)
        f4 ra0 = zero4, ra1 = zero4, rb0 = zero4, rb1 = zero4;
        if (has_next) {
            pa0 += TK; pa1 += TK; pb0 += TK; pb1 += TK;
            if (okA0) { ra0 = *(const f4*)pa0;
                        if (k0 + 2 * TK < K) __builtin_prefetch(pa0 + TK, 0, 3); }
            if (okA1) ra1 = *(const f4*)pa1;
            if (okB0) { rb0 = *(const f4*)pb0;
                        if (k0 + 2 * TK < K) __builtin_prefetch(pb0 + TK, 0, 3); }
            if (okB1) rb1 = *(const f4*)pb1;
        }

        // build per-lane fragments per CDNA5 WMMA VGPR layouts
        h16 a0, a1, b;
        const int ar0  = wr * 32 + lm;
        const int ar1  = ar0 + 16;
        const int brow = wc * 16 + lm;
#pragma unroll
        for (int p = 0; p < 8; ++p) {
            int ka = ((p < 4) ? 2 * p : 2 * p + 8) + hi * 8;  // A 16x32 layout
            a0[2 * p]     = As[cur][ar0][ka];
            a0[2 * p + 1] = As[cur][ar0][ka + 1];
            a1[2 * p]     = As[cur][ar1][ka];
            a1[2 * p + 1] = As[cur][ar1][ka + 1];
            int kb = ((p < 4) ? 2 * p : 2 * p + 8) + hi * 8;  // B 32x16 layout
            b[2 * p]     = Bs[cur][brow][kb];
            b[2 * p + 1] = Bs[cur][brow][kb + 1];
        }
        acc0 = __builtin_amdgcn_wmma_f32_16x16x32_f16(
            false, a0, false, b, (short)0, acc0, false, false);
        acc1 = __builtin_amdgcn_wmma_f32_16x16x32_f16(
            false, a1, false, b, (short)0, acc1, false, false);

        if (has_next) {
            const int nxt = cur ^ 1;
            *(h4*)&As[nxt][sr][sc]      = __builtin_convertvector(ra0, h4);
            *(h4*)&As[nxt][sr + 32][sc] = __builtin_convertvector(ra1, h4);
            *(h4*)&Bs[nxt][sr][sc]      = __builtin_convertvector(rb0, h4);
            *(h4*)&Bs[nxt][sr + 32][sc] = __builtin_convertvector(rb1, h4);
            __syncthreads();            // single barrier per K-step
            cur = nxt;
        }
    }

    // epilogue: C/D layout -> lane holds column lm, rows r + 8*hi
    const int col = n0 + wc * 16 + lm;
    if (col < Nn) {
        const float bv = bias ? bias[col] : 0.f;
#pragma unroll
        for (int r = 0; r < 8; ++r) {
            int row0 = m0 + wr * 32 + r + hi * 8;
            int row1 = row0 + 16;
            float v0 = acc0[r] + bv;
            float v1 = acc1[r] + bv;
            if (act == 1)      { v0 = fmaxf(v0, 0.f); v1 = fmaxf(v1, 0.f); }
            else if (act == 2) {
                v0 = (v0 > 20.f) ? v0 : log1pf(__expf(v0));
                v1 = (v1 > 20.f) ? v1 : log1pf(__expf(v1));
            }
            if (row0 < M) C[(size_t)row0 * ldc + col] = v0;
            if (row1 < M) C[(size_t)row1 * ldc + col] = v1;
        }
    }
}

// =====================================================================
// Instance-norm stats over L per (b, channel) + token construction.
// =====================================================================
__global__ __launch_bounds__(128) void stats_tokens_kernel(
    const float* __restrict__ x_enc,       // (B, L, N)
    const float* __restrict__ x_mark,      // (B, L, MARK)
    float* __restrict__ means,             // (B, N)
    float* __restrict__ stdev,             // (B, N)
    float* __restrict__ tokens)            // (B, T, L)
{
    const int b = blockIdx.x / T_;
    const int t = blockIdx.x % T_;
    const int tid = threadIdx.x;
    float* trow = tokens + ((size_t)b * T_ + t) * L_;

    if (t < N_) {
        __shared__ float s1[128], s2[128];
        float a1 = 0.f, a2 = 0.f;
        for (int l = tid; l < L_; l += 128) {
            float v = x_enc[((size_t)b * L_ + l) * N_ + t];
            a1 += v; a2 += v * v;
        }
        s1[tid] = a1; s2[tid] = a2;
        __syncthreads();
        for (int s = 64; s > 0; s >>= 1) {
            if (tid < s) { s1[tid] += s1[tid + s]; s2[tid] += s2[tid + s]; }
            __syncthreads();
        }
        const float mean = s1[0] * (1.f / L_);
        const float var  = s2[0] * (1.f / L_) - mean * mean;
        const float sd   = sqrtf(var + 1e-5f);
        if (tid == 0) { means[b * N_ + t] = mean; stdev[b * N_ + t] = sd; }
        const float inv = 1.f / sd;
        for (int l = tid; l < L_; l += 128)
            trow[l] = (x_enc[((size_t)b * L_ + l) * N_ + t] - mean) * inv;
    } else {
        const int j = t - N_;
        for (int l = tid; l < L_; l += 128)
            trow[l] = x_mark[((size_t)b * L_ + l) * MARK_ + j];
    }
}

// =====================================================================
// Causal depthwise conv (K=2) + SiLU, float4 over channels
// =====================================================================
__global__ __launch_bounds__(256) void conv_silu_kernel(
    const float* __restrict__ xz,          // (B, T, 2D)
    const float* __restrict__ cw,          // (D, 2)
    const float* __restrict__ cb,          // (D,)
    float* __restrict__ xc)                // (B, T, D)
{
    int idx = blockIdx.x * 256 + threadIdx.x;      // one idx = 4 channels
    if (idx >= (M_ * D_) / 4) return;
    const int bt = idx / (D_ / 4);
    const int d0 = (idx % (D_ / 4)) * 4;
    const int t  = bt % T_;

    f4 xm = *(const f4*)(xz + (size_t)bt * (2 * D_) + d0);
    f4 xp = {0.f, 0.f, 0.f, 0.f};
    if (t > 0) xp = *(const f4*)(xz + (size_t)(bt - 1) * (2 * D_) + d0);

    f4 w01 = *(const f4*)(cw + (size_t)d0 * 2);        // w0[d0],w1[d0],w0[d0+1],w1[d0+1]
    f4 w23 = *(const f4*)(cw + (size_t)d0 * 2 + 4);
    f4 w0 = {w01[0], w01[2], w23[0], w23[2]};
    f4 w1 = {w01[1], w01[3], w23[1], w23[3]};
    f4 bb = *(const f4*)(cb + d0);

    f4 v = xp * w0 + xm * w1 + bb;
    f4 r;
#pragma unroll
    for (int i = 0; i < 4; ++i) r[i] = v[i] / (1.f + __expf(-v[i]));
    *(f4*)(xc + (size_t)bt * D_ + d0) = r;
}

// =====================================================================
// Selective scan: one thread per (b, d); each block owns one batch b
// (256 consecutive d). B/C rows staged in LDS per 32-step chunk.
// Fused with D-skip and SiLU(z) gate.
// =====================================================================
__global__ __launch_bounds__(256) void scan_kernel(
    const float* __restrict__ dt,          // (B, T, D) (post-softplus)
    const float* __restrict__ xc,          // (B, T, D)
    const float* __restrict__ dbl,         // (B, T, 64): [dt(32) | B(16) | C(16)]
    const float* __restrict__ xz,          // (B, T, 2D): z = [:, :, D:]
    const float* __restrict__ a_log,       // (D, S)
    const float* __restrict__ dp,          // (D,)
    float* __restrict__ y)                 // (B, T, D)
{
    __shared__ float sBC[32][32];          // 32 timesteps x [B(16)|C(16)]

    const int idx = blockIdx.x * 256 + threadIdx.x;
    const int b = idx / D_, d = idx % D_;

    float Ae[S_];
#pragma unroll
    for (int s = 0; s < S_; ++s) Ae[s] = -__expf(a_log[d * S_ + s]);

    float hs[S_];
#pragma unroll
    for (int s = 0; s < S_; ++s) hs[s] = 0.f;

    const float dpd = dp[d];
    const float* dblb = dbl + (size_t)b * T_ * 64;

    for (int t0 = 0; t0 < T_; t0 += 32) {
        const int nt = (T_ - t0 < 32) ? (T_ - t0) : 32;
        __syncthreads();
        for (int i = threadIdx.x; i < nt * 32; i += 256) {
            int tt = i >> 5, c = i & 31;
            sBC[tt][c] = dblb[(size_t)(t0 + tt) * 64 + 32 + c];
        }
        __syncthreads();
        for (int tt = 0; tt < nt; ++tt) {
            const size_t bt = (size_t)b * T_ + (t0 + tt);
            const float dtv = dt[bt * D_ + d];
            const float xv  = xc[bt * D_ + d];
            const float* Brow = &sBC[tt][0];
            const float* Crow = &sBC[tt][16];
            float acc = 0.f;
#pragma unroll
            for (int s = 0; s < S_; ++s) {
                hs[s] = __expf(dtv * Ae[s]) * hs[s] + dtv * Brow[s] * xv;
                acc += hs[s] * Crow[s];
            }
            const float zv = xz[bt * (2 * D_) + D_ + d];
            y[bt * D_ + d] = (acc + dpd * xv) * (zv / (1.f + __expf(-zv)));
        }
    }
}

// =====================================================================
// LayerNorm over D=512:  out = LN(a (+ b)) * w + bias  (float4 path)
// =====================================================================
__device__ __forceinline__ float block_reduce_128(float v, float* sh, int tid)
{
    sh[tid] = v; __syncthreads();
    for (int s = 64; s > 0; s >>= 1) {
        if (tid < s) sh[tid] += sh[tid + s];
        __syncthreads();
    }
    float r = sh[0]; __syncthreads();
    return r;
}

__global__ __launch_bounds__(128) void ln_kernel(
    const float* __restrict__ a,
    const float* __restrict__ b,           // may be null
    const float* __restrict__ w,
    const float* __restrict__ bias,
    float* __restrict__ out)
{
    __shared__ float sh[128];
    const int row = blockIdx.x, tid = threadIdx.x;
    const int c0 = tid * 4;

    f4 x = *(const f4*)(a + (size_t)row * D_ + c0);
    if (b) x += *(const f4*)(b + (size_t)row * D_ + c0);

    const float mean = block_reduce_128(x[0] + x[1] + x[2] + x[3], sh, tid) * (1.f / D_);
    f4 dvec = x - mean;
    const float q = dvec[0]*dvec[0] + dvec[1]*dvec[1] + dvec[2]*dvec[2] + dvec[3]*dvec[3];
    const float var = block_reduce_128(q, sh, tid) * (1.f / D_);
    const float rstd = rsqrtf(var + 1e-5f);

    f4 wv = *(const f4*)(w + c0);
    f4 bv = *(const f4*)(bias + c0);
    *(f4*)(out + (size_t)row * D_ + c0) = dvec * rstd * wv + bv;
}

// =====================================================================
// Final: out[b, p, n] = projbuf[b, n, p] * stdev[b,n] + mean[b,n]
// =====================================================================
__global__ __launch_bounds__(256) void out_final_kernel(
    const float* __restrict__ projbuf,     // (B, T, P)
    const float* __restrict__ means,
    const float* __restrict__ stdev,
    float* __restrict__ out)               // (B, P, N)
{
    int idx = blockIdx.x * 256 + threadIdx.x;
    if (idx >= B_ * P_ * N_) return;
    int n = idx % N_;
    int p = (idx / N_) % P_;
    int b = idx / (N_ * P_);
    float v = projbuf[((size_t)b * T_ + n) * P_ + p];
    out[idx] = v * stdev[b * N_ + n] + means[b * N_ + n];
}

// =====================================================================
// host-side orchestration
// =====================================================================
extern "C" void kernel_launch(void* const* d_in, const int* in_sizes, int n_in,
                              void* d_out, int out_size, void* d_ws, size_t ws_size,
                              hipStream_t stream)
{
    (void)in_sizes; (void)n_in; (void)out_size; (void)ws_size;

    const float* x_enc   = (const float*)d_in[0];
    const float* x_mark  = (const float*)d_in[1];
    const float* embed_w = (const float*)d_in[4];
    const float* embed_b = (const float*)d_in[5];
    const float* fw      = (const float*)d_in[23];
    const float* fb      = (const float*)d_in[24];
    const float* proj_w  = (const float*)d_in[25];
    const float* proj_b  = (const float*)d_in[26];
    float* out = (float*)d_out;

    // ---- workspace layout (floats), with aliasing ----
    float* base = (float*)d_ws;
    size_t off = 0;
    float* means  = base + off; off += (size_t)B_ * N_;
    float* stdev  = base + off; off += (size_t)B_ * N_;
    float* tokens = base + off; off += (size_t)M_ * L_;        // alias: mamba-out
    float* hbuf   = base + off; off += (size_t)M_ * D_;
    // region R (dead after scan+out_proj) -> aliased by FFN mid buffer
    float* xzb    = base + off; off += (size_t)M_ * 2 * D_;
    float* xcb    = base + off; off += (size_t)M_ * D_;
    float* dblb   = base + off; off += (size_t)M_ * 64;
    float* dtb    = base + off; off += (size_t)M_ * D_;
    float* ffnb   = xzb;                                       // M_*F_ fits in R
    float* yb     = base + off; off += (size_t)M_ * D_;        // alias: t2
    float* t2b    = yb;
    float* xlnb   = base + off; off += (size_t)M_ * D_;
    float* projb  = base + off; off += (size_t)M_ * P_;
    float* mob    = tokens;

    auto gemm = [&](const float* A, int lda, const float* W, const float* bias,
                    float* C, int ldc, int M, int Nn, int K, int act) {
        dim3 g((Nn + 63) / 64, (M + 63) / 64);
        wmma_gemm_kernel<<<g, 256, 0, stream>>>(A, lda, W, bias, C, ldc, M, Nn, K, act);
    };

    // 1) instance-norm stats + token build
    stats_tokens_kernel<<<B_ * T_, 128, 0, stream>>>(x_enc, x_mark, means, stdev, tokens);

    // 2) embedding GEMM
    gemm(tokens, L_, embed_w, embed_b, hbuf, D_, M_, D_, L_, 0);

    // 3) encoder layers
    for (int l = 0; l < EL_; ++l) {
        const float* in_w  = (const float*)d_in[6]  + (size_t)l * 2 * D_ * D_;
        const float* cw    = (const float*)d_in[7]  + (size_t)l * D_ * 2;
        const float* cb    = (const float*)d_in[8]  + (size_t)l * D_;
        const float* xp_w  = (const float*)d_in[9]  + (size_t)l * (R_ + 2 * S_) * D_;
        const float* dt_w  = (const float*)d_in[10] + (size_t)l * D_ * R_;
        const float* dt_b  = (const float*)d_in[11] + (size_t)l * D_;
        const float* a_log = (const float*)d_in[12] + (size_t)l * D_ * S_;
        const float* dp    = (const float*)d_in[13] + (size_t)l * D_;
        const float* ow    = (const float*)d_in[14] + (size_t)l * D_ * D_;
        const float* n1w   = (const float*)d_in[15] + (size_t)l * D_;
        const float* n1b   = (const float*)d_in[16] + (size_t)l * D_;
        const float* w1    = (const float*)d_in[17] + (size_t)l * F_ * D_;
        const float* b1    = (const float*)d_in[18] + (size_t)l * F_;
        const float* w2    = (const float*)d_in[19] + (size_t)l * D_ * F_;
        const float* b2    = (const float*)d_in[20] + (size_t)l * D_;
        const float* n2w   = (const float*)d_in[21] + (size_t)l * D_;
        const float* n2b   = (const float*)d_in[22] + (size_t)l * D_;

        gemm(hbuf, D_, in_w, nullptr, xzb, 2 * D_, M_, 2 * D_, D_, 0);
        conv_silu_kernel<<<((M_ * D_ / 4) + 255) / 256, 256, 0, stream>>>(xzb, cw, cb, xcb);
        gemm(xcb, D_, xp_w, nullptr, dblb, 64, M_, R_ + 2 * S_, D_, 0);
        gemm(dblb, 64, dt_w, dt_b, dtb, D_, M_, D_, R_, 2);
        scan_kernel<<<(B_ * D_) / 256, 256, 0, stream>>>(
            dtb, xcb, dblb, xzb, a_log, dp, yb);
        gemm(yb, D_, ow, nullptr, mob, D_, M_, D_, D_, 0);
        ln_kernel<<<M_, 128, 0, stream>>>(hbuf, mob, n1w, n1b, xlnb);
        gemm(xlnb, D_, w1, b1, ffnb, F_, M_, F_, D_, 1);
        gemm(ffnb, F_, w2, b2, t2b, D_, M_, D_, F_, 0);
        ln_kernel<<<M_, 128, 0, stream>>>(xlnb, t2b, n2w, n2b, hbuf);
    }

    // 4) final LN
    ln_kernel<<<M_, 128, 0, stream>>>(hbuf, nullptr, fw, fb, xlnb);
    // 5) projection to P=96
    gemm(xlnb, D_, proj_w, proj_b, projb, P_, M_, P_, D_, 0);
    // 6) un-normalize + transpose to (B, P, N)
    out_final_kernel<<<(B_ * P_ * N_ + 255) / 256, 256, 0, stream>>>(
        projb, means, stdev, out);
}
